// SoftmaxAttention_85512798863904
// MI455X (gfx1250) — matrix operand order repaired and checked by reading, not compiled, not run on hardware
//
#include <hip/hip_runtime.h>
#include <hip/hip_fp16.h>

// Flash attention forward, causal, [B=2, H=16, L=2048, D=64] fp32.
// scale = 1/sqrt(head_num) = 0.25, additive mask -1e7 (matches reference).
// Softmax computed in log2 domain: Q pre-scaled by 0.25*log2(e) so exp2f
// maps directly onto v_exp_f32 (hardware exp is base-2).

typedef _Float16 v16h __attribute__((ext_vector_type(16)));
typedef _Float16 h8   __attribute__((ext_vector_type(8)));
typedef float    v8f  __attribute__((ext_vector_type(8)));

#define LQ   2048
#define DHD  64
#define ROWS_PER_WG 128
#define KBLK 32
#define KPAD 72   // 64 + 8 halves  (144B row stride, 16B aligned)
#define VPAD 40   // 32 + 8 halves  (80B row stride, 16B aligned)
#define PPAD 40   // 32 + 8 halves

#define QSCALE   (0.25f * 1.4426950408889634f)   // 1/sqrt(head_num) * log2(e)
#define MASKL2   (-10000000.0f * 1.4426950408889634f)

__device__ __forceinline__ void lds_fence() {
  asm volatile("s_wait_dscnt 0" ::: "memory");
}

__device__ __forceinline__ v16h cat16(h8 a, h8 b) {
  return __builtin_shufflevector(a, b, 0, 1, 2, 3, 4, 5, 6, 7,
                                       8, 9, 10, 11, 12, 13, 14, 15);
}

// ---- DPP16 cross-lane max within each row of 16 lanes (VALU pipe).
template <int CTRL>
__device__ __forceinline__ float dpp_xf(float x) {
  int xi = __builtin_bit_cast(int, x);
  int yi = __builtin_amdgcn_update_dpp(xi, xi, CTRL, 0xF, 0xF, true);
  return __builtin_bit_cast(float, yi);
}
__device__ __forceinline__ float row16_max(float v) {
  v = fmaxf(v, dpp_xf<0xB1>(v));    // quad_perm(1,0,3,2)  xor 1
  v = fmaxf(v, dpp_xf<0x4E>(v));    // quad_perm(2,3,0,1)  xor 2
  v = fmaxf(v, dpp_xf<0x141>(v));   // row_half_mirror     xor 4
  v = fmaxf(v, dpp_xf<0x140>(v));   // row_mirror          xor 8
  return v;
}

__global__ __launch_bounds__(256, 2)
void fa_causal_wmma(const float* __restrict__ Q,
                    const float* __restrict__ K,
                    const float* __restrict__ V,
                    float* __restrict__ O) {
  __shared__ _Float16 Ks[2 * 32 * KPAD];  // double-buffered K block [key][d]
  __shared__ _Float16 Vs[2 * 64 * VPAD];  // double-buffered V^T block [d][key]
  __shared__ _Float16 Ps[8 * 16 * PPAD];  // per-wave P staging, [row][key]

  const int tid  = threadIdx.x;
  const int lane = tid & 31;
  const int wv   = tid >> 5;
  const int l15  = lane & 15;
  const int h    = lane >> 4;            // half-wave id

  const int bh      = blockIdx.x >> 4;               // 32 (b,h) pairs
  const int rowBase = (blockIdx.x & 15) * ROWS_PER_WG;
  const size_t gbase = (size_t)bh * LQ * DHD;

  // ---------------- Q tile load (A-matrix layout, pre-scaled) --------------
  // A 16x32 f16 layout: lane row = l15; lane half h: elems 0..7 -> k=h*8+j,
  // elems 8..15 -> k=h*8+16+(j-8).  aQ[0]: d 0..31, aQ[1]: d 32..63.
  const int r0   = rowBase + wv * 16;
  const int qrow = r0 + l15;
  v16h aQ[2];
  #pragma unroll
  for (int kk = 0; kk < 2; ++kk) {
    #pragma unroll
    for (int g = 0; g < 2; ++g) {
      const int d0 = kk * 32 + h * 8 + g * 16;
      const float4 a = *(const float4*)(Q + gbase + (size_t)qrow * DHD + d0);
      const float4 b = *(const float4*)(Q + gbase + (size_t)qrow * DHD + d0 + 4);
      const float qf[8] = {a.x, a.y, a.z, a.w, b.x, b.y, b.z, b.w};
      #pragma unroll
      for (int j = 0; j < 8; ++j)
        aQ[kk][g * 8 + j] = (_Float16)(qf[j] * QSCALE);
    }
  }

  // Constant ones B-fragment: P x ones(32x16) -> every column = row sum,
  // so the softmax denominator rides the matrix pipe with the same corr
  // rescaling as the output accumulators (no DPP sum chain needed).
  v16h vone;
  #pragma unroll
  for (int j = 0; j < 16; ++j) vone[j] = (_Float16)1.0f;

  // C/D 16x16 f32 layout: vgpr r holds row M = r + 8*h, col N = l15.
  v8f acc[4] = {};                 // O accumulator, 4 x 16-col d tiles
  v8f acc1   = {};                 // running softmax denominator (row sums)
  float mrow[8];
  #pragma unroll
  for (int r = 0; r < 8; ++r) mrow[r] = -1e30f;

  const int nblk = rowBase / KBLK + 4;   // key blocks needed for causality

  // -------- software-pipelined K/V staging: 256 thr x 8 floats each --------
  const int krow = tid >> 3;             // 0..31 key within block
  const int c0   = (tid & 7) * 8;        // 0..56 d offset
  const float* kptr0 = K + gbase + (size_t)krow * DHD + c0;
  const float* vptr0 = V + gbase + (size_t)krow * DHD + c0;
  float4 rka = ((const float4*)kptr0)[0];
  float4 rkb = ((const float4*)kptr0)[1];
  float4 rva = ((const float4*)vptr0)[0];
  float4 rvb = ((const float4*)vptr0)[1];

  for (int kb = 0; kb < nblk; ++kb) {
    const int kbase = kb * KBLK;
    _Float16* KsB = &Ks[(kb & 1) * (32 * KPAD)];
    _Float16* VsB = &Vs[(kb & 1) * (64 * VPAD)];

    // -------- convert staged registers to f16 and store to LDS ------------
    {
      h8 pk;
      pk[0] = (_Float16)rka.x; pk[1] = (_Float16)rka.y;
      pk[2] = (_Float16)rka.z; pk[3] = (_Float16)rka.w;
      pk[4] = (_Float16)rkb.x; pk[5] = (_Float16)rkb.y;
      pk[6] = (_Float16)rkb.z; pk[7] = (_Float16)rkb.w;
      *(h8*)&KsB[krow * KPAD + c0] = pk;
      const float vf[8] = {rva.x, rva.y, rva.z, rva.w,
                           rvb.x, rvb.y, rvb.z, rvb.w};
      #pragma unroll
      for (int j = 0; j < 8; ++j)
        VsB[(c0 + j) * VPAD + krow] = (_Float16)vf[j];  // transpose into LDS
    }
    __syncthreads();   // tiles of block kb visible to all waves

    // -------- issue next block's global loads (land during compute) -------
    if (kb + 1 < nblk) {
      const float* kp = kptr0 + (size_t)(kb + 1) * KBLK * DHD;
      const float* vp = vptr0 + (size_t)(kb + 1) * KBLK * DHD;
      rka = ((const float4*)kp)[0];
      rkb = ((const float4*)kp)[1];
      rva = ((const float4*)vp)[0];
      rvb = ((const float4*)vp)[1];
      if (kb + 2 < nblk) {                              // global_prefetch_b8
        __builtin_prefetch(kp + KBLK * DHD, 0, 1);
        __builtin_prefetch(vp + KBLK * DHD, 0, 1);
      }
    }

    // -------- S = Q * K^T : two 16-key tiles, K-dim = 64 (2 wmma each) -----
    // B 32x16 f16 layout: lane col N = l15; half h elems j<8 -> k = h*8 + j, j>=8 -> k = 16 + h*8 + (j-8).
    v8f S[2] = {};
    #pragma unroll
    for (int n = 0; n < 2; ++n) {
      #pragma unroll
      for (int dh = 0; dh < 2; ++dh) {
        const int a0 = (n * 16 + l15) * KPAD + dh * 32 + h * 8;
        const v16h bk = cat16(*(const h8*)&KsB[a0], *(const h8*)&KsB[a0 + 16]);
        S[n] = __builtin_amdgcn_wmma_f32_16x16x32_f16(
            false, aQ[dh], false, bk, (short)0, S[n], false, false);
      }
    }

    // -------- causal mask: only diagonal blocks need it (wave-uniform) -----
    float s0a[8], s1a[8];
    #pragma unroll
    for (int r = 0; r < 8; ++r) { s0a[r] = S[0][r]; s1a[r] = S[1][r]; }
    if (kbase + KBLK - 1 > r0) {           // some (row,key) pair masked
      const int key0 = kbase + l15;
      #pragma unroll
      for (int r = 0; r < 8; ++r) {
        const int grow = r0 + r + 8 * h;
        s0a[r] += (key0      > grow) ? MASKL2 : 0.0f;
        s1a[r] += (key0 + 16 > grow) ? MASKL2 : 0.0f;
      }
    }

    // -------- online softmax (log2 domain, DPP row-max) --------------------
    // Phase 1: new running maxes + "did any row's max change" wave vote.
    float mnewA[8];
    int chg = 0;
    #pragma unroll
    for (int r = 0; r < 8; ++r) {
      const float mx = row16_max(fmaxf(s0a[r], s1a[r]));
      const float mn = fmaxf(mrow[r], mx);
      mnewA[r] = mn;
      chg |= (mn > mrow[r]) ? 1 : 0;
    }
    // Phase 2: rescale accumulators only if some max moved (corr != 1).
    if (__any(chg)) {
      #pragma unroll
      for (int r = 0; r < 8; ++r) {
        const float corr = exp2f(mrow[r] - mnewA[r]);   // v_exp_f32
        acc1[r] *= corr;
        #pragma unroll
        for (int t = 0; t < 4; ++t) acc[t][r] *= corr;
      }
    }
    // Phase 3: probabilities -> per-wave LDS staging.
    const int pbase = wv * 16 * PPAD;
    #pragma unroll
    for (int r = 0; r < 8; ++r) {
      const float p0 = exp2f(s0a[r] - mnewA[r]);
      const float p1 = exp2f(s1a[r] - mnewA[r]);
      mrow[r] = mnewA[r];
      Ps[pbase + (r + 8 * h) * PPAD + l15]      = (_Float16)p0;
      Ps[pbase + (r + 8 * h) * PPAD + l15 + 16] = (_Float16)p1;
    }
    lds_fence();   // order same-wave cross-lane P store -> load

    // -------- relayout P as A-matrix 16x32 -------------------------------
    const v16h aP = cat16(*(const h8*)&Ps[pbase + l15 * PPAD + h * 16],
                          *(const h8*)&Ps[pbase + l15 * PPAD + h * 16 + 8]);

    // -------- denominator: acc1 += P * ones ------------------------------
    acc1 = __builtin_amdgcn_wmma_f32_16x16x32_f16(
        false, aP, false, vone, (short)0, acc1, false, false);

    // -------- O += P * V : 4 d-tiles of 16 cols, K-dim = 32 keys ----------
    #pragma unroll
    for (int t = 0; t < 4; ++t) {
      const int a0 = (t * 16 + l15) * VPAD + h * 16;
      const v16h bv = cat16(*(const h8*)&VsB[a0], *(const h8*)&VsB[a0 + 8]);
      acc[t] = __builtin_amdgcn_wmma_f32_16x16x32_f16(
          false, aP, false, bv, (short)0, acc[t], false, false);
    }
  }

  // ---------------- epilogue: normalize and store --------------------------
  #pragma unroll
  for (int r = 0; r < 8; ++r) {
    const float inv = 1.0f / acc1[r];   // all columns hold the row sum
    const size_t orow = gbase + (size_t)(r0 + r + 8 * h) * DHD;
    #pragma unroll
    for (int t = 0; t < 4; ++t)
      O[orow + t * 16 + l15] = acc[t][r] * inv;
  }
}

extern "C" void kernel_launch(void* const* d_in, const int* in_sizes, int n_in,
                              void* d_out, int out_size, void* d_ws, size_t ws_size,
                              hipStream_t stream) {
  (void)in_sizes; (void)n_in; (void)d_ws; (void)ws_size; (void)out_size;
  const float* Q = (const float*)d_in[0];
  const float* K = (const float*)d_in[1];
  const float* V = (const float*)d_in[2];
  float* O = (float*)d_out;
  // 32 (b,h) pairs x 16 row-tiles of 128 rows = 512 workgroups, 256 thr (8 waves)
  fa_causal_wmma<<<dim3(512), dim3(256), 0, stream>>>(Q, K, V, O);
}
